// BDHLayer_13048110645549
// MI455X (gfx1250) — compile-verified, matrix operand order checked by reading, not run on hardware
//
#include <hip/hip_runtime.h>

// ---------------- problem constants ----------------
#define T_    2048
#define D_    1024
#define NH_   16
#define N_    1024
#define HN_   16384   // NH_*N_
#define CS_   256
#define HALFC 128
#define NC_   64

// ---------------- WMMA tile config ----------------
#define BM 128
#define BN 128
#define BK 32
#define KSTR 40        // LDS row stride (elements); 80B = multiple of 16B

typedef __bf16 bf16x16 __attribute__((ext_vector_type(16)));
typedef float  f32x8   __attribute__((ext_vector_type(8)));

__device__ __forceinline__ unsigned short f2bf(float f) {
  unsigned int u = __float_as_uint(f);
  u += 0x7FFFu + ((u >> 16) & 1u);   // round-to-nearest-even
  return (unsigned short)(u >> 16);
}
__device__ __forceinline__ float bf2f(unsigned short h) {
  return __uint_as_float(((unsigned int)h) << 16);
}

// Async 16-byte copy global -> LDS (CDNA5 GLOBAL_LOAD_ASYNC_TO_LDS_B128,
// tracked by ASYNCcnt). LDS offset = low 32 bits of the generic shared addr.
__device__ __forceinline__ void async_cp16(const unsigned short* g,
                                           const unsigned short* l) {
  unsigned lds = (unsigned)(unsigned long long)(const void*)l;
  unsigned long long ga = (unsigned long long)g;
  asm volatile("global_load_async_to_lds_b128 %0, %1, off"
               :: "v"(lds), "v"(ga) : "memory");
}
__device__ __forceinline__ void wait_async0() {
  asm volatile("s_wait_asynccnt 0x0" ::: "memory");
}

// Load one 16x32 bf16 fragment from LDS (A or B stored [row][k]).
// Per CDNA5 ISA 7.12.2: lane&15 = row; lane>>4 selects K-half base (0/8);
// element j maps to kk = k0 + j + ((j>>3)<<3).
__device__ __forceinline__ bf16x16 load_frag(const unsigned short* s, int row, int k0l) {
  const unsigned short* p = s + row * KSTR + k0l;
  union U { unsigned int u; __bf16 b[2]; };
  U w[8];
  w[0].u = *(const unsigned int*)(p + 0);
  w[1].u = *(const unsigned int*)(p + 2);
  w[2].u = *(const unsigned int*)(p + 4);
  w[3].u = *(const unsigned int*)(p + 6);
  w[4].u = *(const unsigned int*)(p + 16);
  w[5].u = *(const unsigned int*)(p + 18);
  w[6].u = *(const unsigned int*)(p + 20);
  w[7].u = *(const unsigned int*)(p + 22);
  bf16x16 f;
#pragma unroll
  for (int q = 0; q < 8; ++q) { f[2 * q] = w[q].b[0]; f[2 * q + 1] = w[q].b[1]; }
  return f;
}

// -----------------------------------------------------------------------------
// Templated bf16 WMMA GEMM: out[M,Nout] = A[M,K] . B^T  (inner product over K)
//   BNK = true : B source row-major [n_out][K]  -> async double-buffered pipeline
//   BNK = false: B source row-major [K][n_out]  -> sync staging with transpose
// Epilogues:
//   EPI 0: sqrelu -> bf16 x_sparse [t][hn]
//   EPI 1: causal mask (col<row) -> bf16 scores, per-head cZ; diag-block skip
//   EPI 2: raw f32 yKV [h][t][d]; K loop truncated at rowBase+BM (S is causal)
//   EPI 3: sqrelu * xs(aux bf16) -> bf16 xy [t][hn], head = blockIdx.z
//   EPI 4: raw f32 yMLP [t][d]
// -----------------------------------------------------------------------------
template <int EPI, bool BNK>
__global__ __launch_bounds__(256) void gemm_bf16(
    const unsigned short* __restrict__ Ab, int lda, long aZ,
    const unsigned short* __restrict__ Bb, int ldb, long bZ,
    float* __restrict__ outF, unsigned short* __restrict__ outU, int ldc, long cZ,
    const unsigned short* __restrict__ aux, int K) {
  constexpr int NBUF = BNK ? 2 : 1;
  __shared__ unsigned short sA[NBUF][BM * KSTR];
  __shared__ unsigned short sB[NBUF][BN * KSTR];

  const int tid  = threadIdx.x;
  const int lane = tid & 31;
  const int wave = tid >> 5;
  const int wm = wave >> 1;   // 0..3 -> 32-row strip
  const int wn = wave & 1;    // 0..1 -> 64-col strip
  const int z  = blockIdx.z;
  const int rowBase = blockIdx.y * BM;
  const int nBlock  = blockIdx.x * BN;

  const int hf = lane >> 4;
  const int nn = lane & 15;
  const int k0l = hf * 8;
  const int mr  = nn;

  // ---- fully-masked scores blocks: write zeros, done ----
  if (EPI == 1 && nBlock >= rowBase + BM) {
#pragma unroll
    for (int i = 0; i < 2; ++i)
#pragma unroll
      for (int j = 0; j < 4; ++j)
#pragma unroll
        for (int v = 0; v < 8; ++v) {
          const int gr = rowBase + wm * 32 + i * 16 + v + 8 * hf;
          const int gc = nBlock + wn * 64 + j * 16 + nn;
          outU[(long)z * cZ + (long)gr * ldc + gc] = 0;
        }
    return;
  }

  const unsigned short* A  = Ab + (long)z * aZ + (long)rowBase * lda;
  const unsigned short* Bp = Bb + (long)z * bZ;

  f32x8 acc[2][4];
#pragma unroll
  for (int i = 0; i < 2; ++i)
#pragma unroll
    for (int j = 0; j < 4; ++j)
#pragma unroll
      for (int v = 0; v < 8; ++v) acc[i][j][v] = 0.0f;

  // causal truncation for S@x: S[t,u]==0 for u >= t
  const int Kend = (EPI == 2) ? ((rowBase + BM < K) ? rowBase + BM : K) : K;
  const int nIter = Kend / BK;

  // staging coordinates (each thread moves 16 ushorts of A and of B)
  const int ar = tid >> 1, ac = (tid & 1) * 16;

  auto compute = [&](int buf) {
    bf16x16 af[2], bfg[4];
#pragma unroll
    for (int i = 0; i < 2; ++i)
      af[i] = load_frag(&sA[buf][0], wm * 32 + i * 16 + mr, k0l);
#pragma unroll
    for (int j = 0; j < 4; ++j)
      bfg[j] = load_frag(&sB[buf][0], wn * 64 + j * 16 + mr, k0l);
#pragma unroll
    for (int i = 0; i < 2; ++i)
#pragma unroll
      for (int j = 0; j < 4; ++j)
        acc[i][j] = __builtin_amdgcn_wmma_f32_16x16x32_bf16(
            false, af[i], false, bfg[j], (short)0, acc[i][j], false, false);
  };

  if constexpr (BNK) {
    // ---- async double-buffered pipeline: copy tile k+1 during WMMA on tile k
    auto stage = [&](int k0, int buf) {
      const unsigned short* ga = A + (long)ar * lda + k0 + ac;
      async_cp16(ga,     &sA[buf][ar * KSTR + ac]);
      async_cp16(ga + 8, &sA[buf][ar * KSTR + ac + 8]);
      const unsigned short* gb = Bp + (long)(nBlock + ar) * ldb + k0 + ac;
      async_cp16(gb,     &sB[buf][ar * KSTR + ac]);
      async_cp16(gb + 8, &sB[buf][ar * KSTR + ac + 8]);
    };
    stage(0, 0);
    wait_async0();
    __syncthreads();
    for (int it = 0; it < nIter; ++it) {
      const int cur = it & 1;
      if (it + 1 < nIter) stage((it + 1) * BK, cur ^ 1);
      compute(cur);
      wait_async0();
      __syncthreads();
    }
  } else {
    // ---- synchronous staging (B needs a transpose into sB[n][k]) ----
    for (int it = 0; it < nIter; ++it) {
      const int k0 = it * BK;
      {
        const unsigned short* g = A + (long)ar * lda + k0 + ac;
        uint4 v0 = *(const uint4*)g;
        uint4 v1 = *(const uint4*)(g + 8);
        *(uint4*)&sA[0][ar * KSTR + ac]     = v0;
        *(uint4*)&sA[0][ar * KSTR + ac + 8] = v1;
        if (k0 + BK < Kend) __builtin_prefetch((const void*)(g + BK), 0, 3);
      }
      {
        const int kk = tid >> 3, n0 = (tid & 7) * 16;  // 32 k-rows x 128 n-cols
        const unsigned short* g = Bp + (long)(k0 + kk) * ldb + nBlock + n0;
        union { uint4 v; unsigned short s[8]; } ld0, ld1;
        ld0.v = *(const uint4*)g;
        ld1.v = *(const uint4*)(g + 8);
#pragma unroll
        for (int i = 0; i < 8; ++i) {
          sB[0][(n0 + i) * KSTR + kk]     = ld0.s[i];
          sB[0][(n0 + 8 + i) * KSTR + kk] = ld1.s[i];
        }
        if (k0 + BK < Kend)
          __builtin_prefetch((const void*)(g + (long)BK * ldb), 0, 3);
      }
      __syncthreads();
      compute(0);
      __syncthreads();
    }
  }

  // ---- epilogue ----
#pragma unroll
  for (int i = 0; i < 2; ++i) {
#pragma unroll
    for (int j = 0; j < 4; ++j) {
#pragma unroll
      for (int v = 0; v < 8; ++v) {
        const int gr = rowBase + wm * 32 + i * 16 + v + 8 * hf;  // row
        const int gc = nBlock + wn * 64 + j * 16 + nn;           // col
        float val = acc[i][j][v];
        if (EPI == 0) {                 // sqrelu -> x_sparse bf16 [t][hn]
          float r = fmaxf(val, 0.0f);
          outU[(long)gr * ldc + gc] = f2bf(r * r);
        } else if (EPI == 1) {          // causal mask (u < t) -> scores bf16
          val = (gc < gr) ? val : 0.0f;
          outU[(long)z * cZ + (long)gr * ldc + gc] = f2bf(val);
        } else if (EPI == 2) {          // yKV f32 [h][t][d]
          outF[(long)z * cZ + (long)gr * ldc + gc] = val;
        } else if (EPI == 3) {          // sqrelu * xs -> xy bf16 [t][hn]
          float r = fmaxf(val, 0.0f);
          float ys = r * r;
          float xs = bf2f(aux[(long)gr * HN_ + (long)z * N_ + gc]);
          outU[(long)gr * HN_ + (long)z * N_ + gc] = f2bf(ys * xs);
        } else {                        // EPI 4: yMLP f32 [t][d]
          outF[(long)gr * ldc + gc] = val;
        }
      }
    }
  }
}

// ---------------- pointwise kernels ----------------

__global__ __launch_bounds__(256) void k_cast(const float* __restrict__ in,
                                              unsigned short* __restrict__ out, int n) {
  int i = blockIdx.x * 256 + threadIdx.x;
  if (i < n) out[i] = f2bf(in[i]);
}

__global__ __launch_bounds__(256) void k_rope_tables(float* __restrict__ cosT,
                                                     float* __restrict__ sinT) {
  int gid = blockIdx.x * 256 + threadIdx.x;   // T_ * 128
  int t = gid >> 7, i = gid & 127;
  float idx = 2.0f * (float)i;
  float inv_freq = exp2f(-idx * (1.0f / 16.0f));          // BASE=2^16, ^(idx/256)
  float fr = (float)t * inv_freq;
  float sv = (idx + 0.4f * (float)CS_) * (1.0f / (1.4f * (float)CS_));
  float power = ((float)t - (float)(T_ / 2)) * (1.0f / 512.0f);
  float scale = exp2f(log2f(sv) * power);
  cosT[gid] = cosf(fr) * scale;
  sinT[gid] = sinf(fr) * scale;
}

__global__ __launch_bounds__(256) void k_rope(const unsigned short* __restrict__ xs,
                                              const float* __restrict__ cosT,
                                              const float* __restrict__ sinT,
                                              unsigned short* __restrict__ Q) {
  int gid = blockIdx.x * 256 + threadIdx.x;   // T_ * NC_ * 128
  int t = gid >> 13;
  int rem = gid & 8191;
  int c = rem >> 7, j = rem & 127;
  long base = (long)t * HN_ + c * CS_;
  float q1 = bf2f(xs[base + j]);
  float q2 = bf2f(xs[base + HALFC + j]);
  float cs = cosT[t * 128 + j], sn = sinT[t * 128 + j];
  float r1 = q1 * cs - q2 * sn;
  float r2 = q2 * cs + q1 * sn;
  int hn1 = c * CS_ + j, hn2 = hn1 + HALFC;
  Q[(long)(hn1 >> 10) * T_ * N_ + (long)t * N_ + (hn1 & 1023)] = f2bf(r1);
  Q[(long)(hn2 >> 10) * T_ * N_ + (long)t * N_ + (hn2 & 1023)] = f2bf(r2);
}

__device__ __forceinline__ void block_sum2(float& a, float& b, float* s1, float* s2) {
  int tid = threadIdx.x;
  s1[tid] = a; s2[tid] = b;
  __syncthreads();
  for (int s = 128; s > 0; s >>= 1) {
    if (tid < s) { s1[tid] += s1[tid + s]; s2[tid] += s2[tid + s]; }
    __syncthreads();
  }
  a = s1[0]; b = s2[0];
  __syncthreads();
}

__global__ __launch_bounds__(256) void k_ln(const float* __restrict__ yKV,
                                            unsigned short* __restrict__ out) {
  __shared__ float s1[256], s2[256];
  const long row = blockIdx.x;                 // h*T_ + t
  const float* p = yKV + row * D_;
  float v[4], sum = 0.0f, sq = 0.0f;
#pragma unroll
  for (int i = 0; i < 4; ++i) {
    v[i] = p[threadIdx.x + i * 256];
    sum += v[i]; sq += v[i] * v[i];
  }
  block_sum2(sum, sq, s1, s2);
  float mu = sum * (1.0f / D_);
  float var = sq * (1.0f / D_) - mu * mu;
  float rs = rsqrtf(var + 1e-5f);
#pragma unroll
  for (int i = 0; i < 4; ++i)
    out[row * D_ + threadIdx.x + i * 256] = f2bf((v[i] - mu) * rs);
}

__global__ __launch_bounds__(256) void k_final(const float* __restrict__ yMLP,
                                               const float* __restrict__ x,
                                               float* __restrict__ out) {
  __shared__ float s1[256], s2[256];
  const long t = blockIdx.x;
  const float* p  = yMLP + t * D_;
  const float* xr = x + t * D_;
  float v[4], sum = 0.0f, sq = 0.0f;
#pragma unroll
  for (int i = 0; i < 4; ++i) {
    v[i] = p[threadIdx.x + i * 256];
    sum += v[i]; sq += v[i] * v[i];
  }
  block_sum2(sum, sq, s1, s2);
  float mu = sum * (1.0f / D_);
  float var = sq * (1.0f / D_) - mu * mu;
  float rs = rsqrtf(var + 1e-5f);
  float zz[4], s2n = 0.0f, dummy = 0.0f;
#pragma unroll
  for (int i = 0; i < 4; ++i) {
    zz[i] = (v[i] - mu) * rs + xr[threadIdx.x + i * 256];
    s2n += zz[i] * zz[i];
  }
  block_sum2(s2n, dummy, s1, s2);
  float r2 = rsqrtf(s2n * (1.0f / D_) + 1e-6f);
#pragma unroll
  for (int i = 0; i < 4; ++i)
    out[t * D_ + threadIdx.x + i * 256] = zz[i] * r2;
}

// ---------------- host-side orchestration ----------------
extern "C" void kernel_launch(void* const* d_in, const int* in_sizes, int n_in,
                              void* d_out, int out_size, void* d_ws, size_t ws_size,
                              hipStream_t stream) {
  const float* x      = (const float*)d_in[0];
  const float* W_enc  = (const float*)d_in[1];
  const float* W_encv = (const float*)d_in[2];
  const float* W_dec  = (const float*)d_in[3];
  float* out = (float*)d_out;

  char* w = (char*)d_ws;
  size_t o = 0;
  auto alloc = [&](size_t bytes) -> void* {
    void* p = w + o;
    o = (o + bytes + 255) & ~(size_t)255;
    return p;
  };

  unsigned short* xb     = (unsigned short*)alloc((size_t)T_ * D_ * 2);
  unsigned short* Wenc_b = (unsigned short*)alloc((size_t)HN_ * D_ * 2);
  unsigned short* Wenv_b = (unsigned short*)alloc((size_t)HN_ * D_ * 2);
  unsigned short* Wdec_b = (unsigned short*)alloc((size_t)D_ * HN_ * 2);
  float* cosT            = (float*)alloc((size_t)T_ * 128 * 4);
  float* sinT            = (float*)alloc((size_t)T_ * 128 * 4);
  unsigned short* xs     = (unsigned short*)alloc((size_t)T_ * HN_ * 2);
  unsigned short* Q      = (unsigned short*)alloc((size_t)NH_ * T_ * N_ * 2);
  unsigned short* S      = (unsigned short*)alloc((size_t)NH_ * T_ * T_ * 2);
  float* yKV             = (float*)alloc((size_t)NH_ * T_ * D_ * 4);
  // lifetime-based aliases
  unsigned short* yKVn = Q;                 // written after Q is dead
  unsigned short* xy   = S;                 // written after S is dead
  float* yMLP          = yKV;               // written after yKV is dead

  // 1) casts to bf16
  k_cast<<<(T_ * D_) / 256, 256, 0, stream>>>(x, xb, T_ * D_);
  k_cast<<<(HN_ * D_) / 256, 256, 0, stream>>>(W_enc, Wenc_b, HN_ * D_);
  k_cast<<<(HN_ * D_) / 256, 256, 0, stream>>>(W_encv, Wenv_b, HN_ * D_);
  k_cast<<<(D_ * HN_) / 256, 256, 0, stream>>>(W_dec, Wdec_b, D_ * HN_);

  // 2) RoPE tables
  k_rope_tables<<<(T_ * 128) / 256, 256, 0, stream>>>(cosT, sinT);

  // 3) encode: xs = sqrelu(x @ W_enc^T)   [T, HN]
  gemm_bf16<0, true><<<dim3(HN_ / BN, T_ / BM, 1), 256, 0, stream>>>(
      xb, D_, 0L, Wenc_b, D_, 0L, nullptr, xs, HN_, 0L, nullptr, D_);

  // 4) RoPE -> Q  [h][t][n] bf16
  k_rope<<<(T_ * NC_ * 128) / 256, 256, 0, stream>>>(xs, cosT, sinT, Q);

  // 5) scores: S_h = tril(Q_h Q_h^T, -1)  [h][t][u] bf16
  gemm_bf16<1, true><<<dim3(T_ / BN, T_ / BM, NH_), 256, 0, stream>>>(
      Q, N_, (long)T_ * N_, Q, N_, (long)T_ * N_,
      nullptr, S, T_, (long)T_ * T_, nullptr, N_);

  // 6) yKV_h = S_h @ x   [h][t][d] f32   (B natural: x[u][d])
  gemm_bf16<2, false><<<dim3(D_ / BN, T_ / BM, NH_), 256, 0, stream>>>(
      S, T_, (long)T_ * T_, xb, D_, 0L,
      yKV, nullptr, D_, (long)T_ * D_, nullptr, T_);

  // 7) layernorm over d -> bf16 [h][t][d]
  k_ln<<<NH_ * T_, 256, 0, stream>>>(yKV, yKVn);

  // 8) encode_v + fuse xy = x_sparse * sqrelu(yKVn @ W_enc_v^T) -> xy bf16 [t][hn]
  gemm_bf16<3, true><<<dim3(N_ / BN, T_ / BM, NH_), 256, 0, stream>>>(
      yKVn, D_, (long)T_ * D_, Wenv_b, D_, (long)N_ * D_,
      nullptr, xy, HN_, 0L, xs, D_);

  // 9) decode: yMLP = xy @ W_dec^T  [t][d] f32
  gemm_bf16<4, true><<<dim3(D_ / BN, T_ / BM, 1), 256, 0, stream>>>(
      xy, HN_, 0L, Wdec_b, HN_, 0L, yMLP, nullptr, D_, 0L, nullptr, HN_);

  // 10) out = rmsnorm(layernorm(yMLP) + x)
  k_final<<<T_, 256, 0, stream>>>(yMLP, x, out);

  (void)in_sizes; (void)n_in; (void)out_size; (void)ws_size;
}
